// AuxiliaryGAT_84670985273384
// MI455X (gfx1250) — matrix operand order repaired and checked by reading, not compiled, not run on hardware
//
#include <hip/hip_runtime.h>
#include <hip/hip_bf16.h>

// ---------------- problem constants ----------------
constexpr int NN   = 50000;    // nodes
constexpr int EE   = 800000;   // edges
constexpr int EP   = EE + NN;  // edges + self loops
constexpr int INC  = 128;
constexpr int HID  = 32;
constexpr int HEADS= 8;
constexpr int HW   = HID * HEADS; // 256
constexpr int OUTC = 40;
constexpr float NEG_SLOPE = 0.2f;
constexpr float BN_EPS = 1e-5f;

#define CDIV(a,b) (((a)+(b)-1)/(b))

typedef __bf16 bf16_t;
typedef __attribute__((ext_vector_type(16))) __bf16 v16bf;
typedef __attribute__((ext_vector_type(8)))  __bf16 v8bf;
typedef __attribute__((ext_vector_type(8)))  float  v8f;

union BF16x16 { v16bf v; v8bf h[2]; };

// ---------------- helpers ----------------
__device__ __forceinline__ int f2ord(float f) {
    int b = __float_as_int(f);
    return b >= 0 ? b : (b ^ 0x7fffffff);
}
__device__ __forceinline__ float ord2f(int k) {
    int b = k >= 0 ? k : (k ^ 0x7fffffff);
    return __int_as_float(b);
}
__device__ __forceinline__ float lrelu(float x) { return x > 0.f ? x : NEG_SLOPE * x; }

__device__ __forceinline__ void edge_sd(const int* __restrict__ ei, int e, int& s, int& d) {
    if (e < EE) { s = ei[e]; d = ei[EE + e]; }
    else        { s = e - EE; d = e - EE; }   // self loop
}

// ---------------- packing ----------------
__global__ void k_f32_to_bf16(const float* __restrict__ in, bf16_t* __restrict__ out, int n) {
    int i = blockIdx.x * blockDim.x + threadIdx.x;
    if (i < n) out[i] = (bf16_t)in[i];
}

// W [K][C] row-major (f32) -> Wt [C][K] (bf16) so B fragments are K-contiguous per column
__global__ void k_pack_wt(const float* __restrict__ W, bf16_t* __restrict__ Wt, int K, int C) {
    int i = blockIdx.x * blockDim.x + threadIdx.x;
    if (i >= K * C) return;
    int k = i / C, c = i % C;
    Wt[c * K + k] = (bf16_t)W[i];
}

// ---------------- WMMA GEMMs ----------------
// h1[N,256] = xb[N,128](bf16) @ W1t(colmajor bf16).
// One wave computes a 16x64 strip with 4 independent accumulators: no WMMA->WMMA
// RAW hazard NOPs, and the A fragment is loaded once per K-step (4x A reuse).
__global__ __launch_bounds__(256) void k_gemm1(const bf16_t* __restrict__ xb,
                                               const bf16_t* __restrict__ w1t,
                                               float* __restrict__ h1) {
    int wave = (blockIdx.x * blockDim.x + threadIdx.x) >> 5;   // 0 .. 3125*4-1
    if (wave >= (NN / 16) * 4) return;                          // wave-uniform
    int lane = threadIdx.x & 31;
    int mtile = wave >> 2, nquad = wave & 3;                    // cols [nquad*64, +64)
    int l15 = lane & 15, khalf = lane >> 4;
    const bf16_t* arow  = xb  + (size_t)(mtile * 16 + l15) * INC;
    const bf16_t* bbase = w1t + (size_t)(nquad * 64 + l15) * INC;
    v8f acc[4] = {};
#pragma unroll
    for (int k0 = 0; k0 < INC; k0 += 32) {
        BF16x16 A;
        A.h[0] = *(const v8bf*)(arow + k0 + khalf * 8);
        A.h[1] = *(const v8bf*)(arow + k0 + 16 + khalf * 8);
#pragma unroll
        for (int t = 0; t < 4; ++t) {
            BF16x16 B;
            const bf16_t* bcol = bbase + (size_t)(t * 16) * INC;
            B.h[0] = *(const v8bf*)(bcol + k0 + khalf * 8);
            B.h[1] = *(const v8bf*)(bcol + k0 + 16 + khalf * 8);
            acc[t] = __builtin_amdgcn_wmma_f32_16x16x32_bf16(false, A.v, false, B.v,
                                                             (short)0, acc[t], false, false);
        }
    }
#pragma unroll
    for (int t = 0; t < 4; ++t) {
        float* out = h1 + (size_t)(mtile * 16 + khalf * 8) * HW + nquad * 64 + t * 16 + l15;
#pragma unroll
        for (int r = 0; r < 8; ++r) out[(size_t)r * HW] = acc[t][r];
    }
}

// h2[N,32] = hb[N,256](bf16) @ W2t(colmajor bf16); one wave = full 16x32 strip.
__global__ __launch_bounds__(256) void k_gemm2(const bf16_t* __restrict__ hb,
                                               const bf16_t* __restrict__ w2t,
                                               float* __restrict__ h2) {
    int wave = (blockIdx.x * blockDim.x + threadIdx.x) >> 5;
    if (wave >= NN / 16) return;                                // wave-uniform
    int lane = threadIdx.x & 31;
    int mtile = wave;
    int l15 = lane & 15, khalf = lane >> 4;
    const bf16_t* arow  = hb  + (size_t)(mtile * 16 + l15) * HW;
    const bf16_t* bbase = w2t + (size_t)l15 * HW;
    v8f acc[2] = {};
#pragma unroll
    for (int k0 = 0; k0 < HW; k0 += 32) {
        BF16x16 A;
        A.h[0] = *(const v8bf*)(arow + k0 + khalf * 8);
        A.h[1] = *(const v8bf*)(arow + k0 + 16 + khalf * 8);
#pragma unroll
        for (int t = 0; t < 2; ++t) {
            BF16x16 B;
            const bf16_t* bcol = bbase + (size_t)(t * 16) * HW;
            B.h[0] = *(const v8bf*)(bcol + k0 + khalf * 8);
            B.h[1] = *(const v8bf*)(bcol + k0 + 16 + khalf * 8);
            acc[t] = __builtin_amdgcn_wmma_f32_16x16x32_bf16(false, A.v, false, B.v,
                                                             (short)0, acc[t], false, false);
        }
    }
#pragma unroll
    for (int t = 0; t < 2; ++t) {
        float* out = h2 + (size_t)(mtile * 16 + khalf * 8) * HID + t * 16 + l15;
#pragma unroll
        for (int r = 0; r < 8; ++r) out[(size_t)r * HID] = acc[t][r];
    }
}

// ---------------- attention layer 1 (8 heads) ----------------
__global__ void k_alpha1(const float* __restrict__ h1, const float* __restrict__ aS,
                         const float* __restrict__ aD, float* __restrict__ oS,
                         float* __restrict__ oD) {
    int t = blockIdx.x * blockDim.x + threadIdx.x;
    if (t >= NN * HEADS) return;
    int n = t >> 3, h = t & 7;
    const float* hp = h1 + (size_t)n * HW + h * HID;
    const float* as = aS + h * HID;
    const float* ad = aD + h * HID;
    float s = 0.f, d = 0.f;
#pragma unroll
    for (int c = 0; c < HID; ++c) { float v = hp[c]; s += v * as[c]; d += v * ad[c]; }
    oS[t] = s; oD[t] = d;
}

__global__ void k_init1(float* __restrict__ out1, const float* __restrict__ b1,
                        int* __restrict__ mx, float* __restrict__ den) {
    int i = blockIdx.x * blockDim.x + threadIdx.x;
    if (i < NN * HW) out1[i] = b1[i & (HW - 1)];
    if (i < NN * HEADS) { mx[i] = INT_MIN; den[i] = 0.f; }
}

__global__ void k_logits_max1(const int* __restrict__ ei, const float* __restrict__ aS,
                              const float* __restrict__ aD, float* __restrict__ e1,
                              int* __restrict__ mx) {
    int t = blockIdx.x * blockDim.x + threadIdx.x;
    if (t >= EP * HEADS) return;
    int e = t >> 3, h = t & 7, s, d;
    edge_sd(ei, e, s, d);
    float l = lrelu(aS[s * HEADS + h] + aD[d * HEADS + h]);
    e1[t] = l;
    atomicMax(&mx[d * HEADS + h], f2ord(l));
}

__global__ void k_exp_den1(const int* __restrict__ ei, float* __restrict__ e1,
                           const int* __restrict__ mx, float* __restrict__ den) {
    int t = blockIdx.x * blockDim.x + threadIdx.x;
    if (t >= EP * HEADS) return;
    int e = t >> 3, h = t & 7, s, d;
    edge_sd(ei, e, s, d);
    float ev = __expf(e1[t] - ord2f(mx[d * HEADS + h]));
    e1[t] = ev;
    atomicAdd(&den[d * HEADS + h], ev);
}

__global__ void k_agg1(const int* __restrict__ ei, const float* __restrict__ h1,
                       const float* __restrict__ e1, const float* __restrict__ den,
                       float* __restrict__ out1) {
    int t = blockIdx.x * blockDim.x + threadIdx.x;
    if (t >= EP * 32) return;
    int e = t >> 5, lane = t & 31, head = lane >> 2, s, d;
    edge_sd(ei, e, s, d);
    float alpha = e1[e * HEADS + head] / (den[d * HEADS + head] + 1e-16f);
    const float* hs = h1 + (size_t)s * HW + lane * 8;
    float* od = out1 + (size_t)d * HW + lane * 8;
#pragma unroll
    for (int j = 0; j < 8; ++j) atomicAdd(od + j, hs[j] * alpha);
}

// BN (eval) + ELU, write bf16 for GEMM2
__global__ void k_bnelu(const float* __restrict__ in, const float* __restrict__ g,
                        const float* __restrict__ b, const float* __restrict__ m,
                        const float* __restrict__ v, bf16_t* __restrict__ out) {
    int i = blockIdx.x * blockDim.x + threadIdx.x;
    if (i >= NN * HW) return;
    int c = i & (HW - 1);
    float sc = g[c] * rsqrtf(v[c] + BN_EPS);
    float x = (in[i] - m[c]) * sc + b[c];
    x = x > 0.f ? x : (__expf(x) - 1.f);
    out[i] = (bf16_t)x;
}

// ---------------- attention layer 2 (1 head) ----------------
__global__ void k_alpha2(const float* __restrict__ h2, const float* __restrict__ aS,
                         const float* __restrict__ aD, float* __restrict__ oS,
                         float* __restrict__ oD) {
    int n = blockIdx.x * blockDim.x + threadIdx.x;
    if (n >= NN) return;
    const float* hp = h2 + (size_t)n * HID;
    float s = 0.f, d = 0.f;
#pragma unroll
    for (int c = 0; c < HID; ++c) { float v = hp[c]; s += v * aS[c]; d += v * aD[c]; }
    oS[n] = s; oD[n] = d;
}

__global__ void k_init2(float* __restrict__ out2, const float* __restrict__ b2,
                        int* __restrict__ mx, float* __restrict__ den) {
    int i = blockIdx.x * blockDim.x + threadIdx.x;
    if (i < NN * HID) out2[i] = b2[i & (HID - 1)];
    if (i < NN) { mx[i] = INT_MIN; den[i] = 0.f; }
}

__global__ void k_logits_max2(const int* __restrict__ ei, const float* __restrict__ aS,
                              const float* __restrict__ aD, float* __restrict__ e2,
                              int* __restrict__ mx) {
    int e = blockIdx.x * blockDim.x + threadIdx.x;
    if (e >= EP) return;
    int s, d; edge_sd(ei, e, s, d);
    float l = lrelu(aS[s] + aD[d]);
    e2[e] = l;
    atomicMax(&mx[d], f2ord(l));
}

__global__ void k_exp_den2(const int* __restrict__ ei, float* __restrict__ e2,
                           const int* __restrict__ mx, float* __restrict__ den) {
    int e = blockIdx.x * blockDim.x + threadIdx.x;
    if (e >= EP) return;
    int s, d; edge_sd(ei, e, s, d);
    float ev = __expf(e2[e] - ord2f(mx[d]));
    e2[e] = ev;
    atomicAdd(&den[d], ev);
}

__global__ void k_agg2(const int* __restrict__ ei, const float* __restrict__ h2,
                       const float* __restrict__ e2, const float* __restrict__ den,
                       float* __restrict__ out2) {
    int t = blockIdx.x * blockDim.x + threadIdx.x;
    if (t >= EP * HID) return;
    int e = t >> 5, c = t & 31, s, d;
    edge_sd(ei, e, s, d);
    float alpha = e2[e] / (den[d] + 1e-16f);
    atomicAdd(&out2[(size_t)d * HID + c], h2[(size_t)s * HID + c] * alpha);
}

// ---------------- fused output heads ----------------
__global__ __launch_bounds__(256) void k_heads(const float* __restrict__ out2,
        const float* __restrict__ Wc, const float* __restrict__ bc,
        const float* __restrict__ Ws, const float* __restrict__ bs,
        const float* __restrict__ Wh, const float* __restrict__ bh,
        const float* __restrict__ We, const float* __restrict__ be,
        float* __restrict__ mainO, float* __restrict__ simO,
        float* __restrict__ homO, float* __restrict__ entO) {
    __shared__ float sWc[HID * OUTC], sWs[HID * OUTC];
    __shared__ float sbc[OUTC], sbs[OUTC], sWh[HID], sWe[HID];
    for (int i = threadIdx.x; i < HID * OUTC; i += blockDim.x) { sWc[i] = Wc[i]; sWs[i] = Ws[i]; }
    if (threadIdx.x < OUTC) { sbc[threadIdx.x] = bc[threadIdx.x]; sbs[threadIdx.x] = bs[threadIdx.x]; }
    if (threadIdx.x < HID)  { sWh[threadIdx.x] = Wh[threadIdx.x]; sWe[threadIdx.x] = We[threadIdx.x]; }
    __syncthreads();

    int n = blockIdx.x * blockDim.x + threadIdx.x;
    if (n >= NN) return;
    float h[HID];
#pragma unroll
    for (int c = 0; c < HID; ++c) h[c] = out2[(size_t)n * HID + c];

    float z[OUTC];
    // main: log_softmax(h@Wc + bc)
    float mx = -3.402823466e38f;
    for (int j = 0; j < OUTC; ++j) {
        float s = sbc[j];
#pragma unroll
        for (int c = 0; c < HID; ++c) s += h[c] * sWc[c * OUTC + j];
        z[j] = s; mx = fmaxf(mx, s);
    }
    float sum = 0.f;
    for (int j = 0; j < OUTC; ++j) sum += __expf(z[j] - mx);
    float lse = mx + __logf(sum);
    for (int j = 0; j < OUTC; ++j) mainO[(size_t)n * OUTC + j] = z[j] - lse;

    // sim: softmax(h@Ws + bs)
    mx = -3.402823466e38f;
    for (int j = 0; j < OUTC; ++j) {
        float s = sbs[j];
#pragma unroll
        for (int c = 0; c < HID; ++c) s += h[c] * sWs[c * OUTC + j];
        z[j] = s; mx = fmaxf(mx, s);
    }
    sum = 0.f;
    for (int j = 0; j < OUTC; ++j) { z[j] = __expf(z[j] - mx); sum += z[j]; }
    float inv = 1.f / sum;
    for (int j = 0; j < OUTC; ++j) simO[(size_t)n * OUTC + j] = z[j] * inv;

    // hom / ent sigmoids
    float sh = bh[0], se = be[0];
#pragma unroll
    for (int c = 0; c < HID; ++c) { sh += h[c] * sWh[c]; se += h[c] * sWe[c]; }
    homO[n] = 1.f / (1.f + __expf(-sh));
    entO[n] = 1.f / (1.f + __expf(-se));
}

// ---------------- launcher ----------------
extern "C" void kernel_launch(void* const* d_in, const int* in_sizes, int n_in,
                              void* d_out, int out_size, void* d_ws, size_t ws_size,
                              hipStream_t stream) {
    (void)in_sizes; (void)n_in; (void)out_size; (void)ws_size;
    const float* x    = (const float*)d_in[0];
    const int*   ei   = (const int*)  d_in[1];
    const float* W1   = (const float*)d_in[2];
    const float* aS1  = (const float*)d_in[3];
    const float* aD1  = (const float*)d_in[4];
    const float* b1   = (const float*)d_in[5];
    const float* bn_g = (const float*)d_in[6];
    const float* bn_b = (const float*)d_in[7];
    const float* bn_m = (const float*)d_in[8];
    const float* bn_v = (const float*)d_in[9];
    const float* W2   = (const float*)d_in[10];
    const float* aS2  = (const float*)d_in[11];
    const float* aD2  = (const float*)d_in[12];
    const float* b2   = (const float*)d_in[13];
    const float* Wc   = (const float*)d_in[14];
    const float* bc   = (const float*)d_in[15];
    const float* Ws   = (const float*)d_in[16];
    const float* bs   = (const float*)d_in[17];
    const float* Wh   = (const float*)d_in[18];
    const float* bh   = (const float*)d_in[19];
    const float* We   = (const float*)d_in[20];
    const float* be   = (const float*)d_in[21];

    float* mainO = (float*)d_out;
    float* simO  = mainO + (size_t)NN * OUTC;
    float* homO  = simO  + (size_t)NN * OUTC;
    float* entO  = homO  + NN;

    char* ws = (char*)d_ws;
    size_t off = 0;
    auto alloc = [&](size_t bytes) -> void* {
        void* p = ws + off;
        off += (bytes + 255) & ~(size_t)255;
        return p;
    };
    bf16_t* xb   = (bf16_t*)alloc((size_t)NN * INC * 2);
    bf16_t* w1t  = (bf16_t*)alloc((size_t)HW * INC * 2);
    bf16_t* w2t  = (bf16_t*)alloc((size_t)HID * HW * 2);
    float*  h1   = (float*) alloc((size_t)NN * HW * 4);
    float*  aS1v = (float*) alloc((size_t)NN * HEADS * 4);
    float*  aD1v = (float*) alloc((size_t)NN * HEADS * 4);
    int*    max1 = (int*)   alloc((size_t)NN * HEADS * 4);
    float*  den1 = (float*) alloc((size_t)NN * HEADS * 4);
    float*  e1   = (float*) alloc((size_t)EP * HEADS * 4);
    float*  out1 = (float*) alloc((size_t)NN * HW * 4);
    bf16_t* hb   = (bf16_t*)alloc((size_t)NN * HW * 2);
    float*  h2   = (float*) alloc((size_t)NN * HID * 4);
    float*  aS2v = (float*) alloc((size_t)NN * 4);
    float*  aD2v = (float*) alloc((size_t)NN * 4);
    int*    max2 = (int*)   alloc((size_t)NN * 4);
    float*  den2 = (float*) alloc((size_t)NN * 4);
    float*  e2   = (float*) alloc((size_t)EP * 4);
    float*  out2 = (float*) alloc((size_t)NN * HID * 4);

    const int B = 256;
    // pack
    k_f32_to_bf16<<<CDIV(NN * INC, B), B, 0, stream>>>(x, xb, NN * INC);
    k_pack_wt<<<CDIV(INC * HW, B), B, 0, stream>>>(W1, w1t, INC, HW);
    k_pack_wt<<<CDIV(HW * HID, B), B, 0, stream>>>(W2, w2t, HW, HID);
    // layer 1
    k_gemm1<<<CDIV((NN / 16) * 4, 8), B, 0, stream>>>(xb, w1t, h1);
    k_alpha1<<<CDIV(NN * HEADS, B), B, 0, stream>>>(h1, aS1, aD1, aS1v, aD1v);
    k_init1<<<CDIV(NN * HW, B), B, 0, stream>>>(out1, b1, max1, den1);
    k_logits_max1<<<CDIV(EP * HEADS, B), B, 0, stream>>>(ei, aS1v, aD1v, e1, max1);
    k_exp_den1<<<CDIV(EP * HEADS, B), B, 0, stream>>>(ei, e1, max1, den1);
    k_agg1<<<CDIV(EP * 32, B), B, 0, stream>>>(ei, h1, e1, den1, out1);
    k_bnelu<<<CDIV(NN * HW, B), B, 0, stream>>>(out1, bn_g, bn_b, bn_m, bn_v, hb);
    // layer 2
    k_gemm2<<<CDIV(NN / 16, 8), B, 0, stream>>>(hb, w2t, h2);
    k_alpha2<<<CDIV(NN, B), B, 0, stream>>>(h2, aS2, aD2, aS2v, aD2v);
    k_init2<<<CDIV(NN * HID, B), B, 0, stream>>>(out2, b2, max2, den2);
    k_logits_max2<<<CDIV(EP, B), B, 0, stream>>>(ei, aS2v, aD2v, e2, max2);
    k_exp_den2<<<CDIV(EP, B), B, 0, stream>>>(ei, e2, max2, den2);
    k_agg2<<<CDIV(EP * HID, B), B, 0, stream>>>(ei, h2, e2, den2, out2);
    // heads
    k_heads<<<CDIV(NN, B), B, 0, stream>>>(out2, Wc, bc, Ws, bs, Wh, bh, We, be,
                                           mainO, simO, homO, entO);
}
